// Block_38732015076044
// MI455X (gfx1250) — matrix operand order repaired and checked by reading, not compiled, not run on hardware
//
#include <hip/hip_runtime.h>
#include <hip/hip_bf16.h>
#include <math.h>
#include <stdint.h>

// ---------- problem sizes ----------
#define DIM   512
#define NKEY  128
#define BQ    512
#define BKB   256
#define NC    1000
#define HID   2048
#define NPAD  1024   // NC padded to multiple of 64 for the GEMM
#define KC    128    // K-chunk staged in LDS by the TDM per GEMM block

typedef __attribute__((ext_vector_type(16))) __bf16 v16bf;
typedef __attribute__((ext_vector_type(8)))  float  v8f;
typedef __attribute__((ext_vector_type(4)))  unsigned int u32x4;
typedef __attribute__((ext_vector_type(8)))  int i32x8;
typedef __attribute__((ext_vector_type(4)))  int i32x4;

#if __has_builtin(__builtin_amdgcn_tensor_load_to_lds)
#define USE_TDM 1
#else
#define USE_TDM 0
#endif

// ---------------- WMMA fragment loaders (wave32, gfx1250 layouts) -------------
// A-matrix 16x32 bf16: lane L(0-15) row M=L holds K=0..15 contiguous;
// lane L+16 holds K=16..31 (same k order as load_b_frag).
static __device__ __forceinline__ v16bf load_a_frag(const __bf16* base, int ld) {
  int lane = threadIdx.x & 31;
  int r    = lane & 15;
  int koff = (lane >> 4) << 4;           // 0 or 16
  const __bf16* p = base + (size_t)r * ld + koff;
  v16bf f;
  ((float4*)&f)[0] = *(const float4*)(p);        // K = koff .. koff+7
  ((float4*)&f)[1] = *(const float4*)(p + 8);    // K = koff+8 .. koff+15
  return f;
}
// B-matrix 32x16 bf16 stored as Bt[N][K] row-major: lane L(0-15) col N=L holds
// K=0..15 contiguous; lane L+16 holds K=16..31. (same k order as load_a_frag)
static __device__ __forceinline__ v16bf load_b_frag(const __bf16* base, int ld) {
  int lane = threadIdx.x & 31;
  int c    = lane & 15;
  int koff = (lane >> 4) << 4;           // 0 or 16
  const __bf16* p = base + (size_t)c * ld + koff;
  v16bf f;
  ((float4*)&f)[0] = *(const float4*)(p);
  ((float4*)&f)[1] = *(const float4*)(p + 8);
  return f;
}

#if USE_TDM
// ---------------- Tensor Data Mover: 2D tile (rows x cols bf16) -> LDS --------
// D# packed per CDNA5 ISA ch.8: group0 {count=1, lds_addr, global_addr, type=2},
// group1 {data_size=2B, tensor dims (huge: tile always in-bounds), tile dims,
// tensor_dim0_stride = row stride}.  Tracked on TENSORcnt.
// This toolchain's builtin is the 6-arg form:
//   (u32x4 g0, i32x8 g1, i32x4 g2, i32x4 g3, i32x8 extra, i32 cpol)
static __device__ __forceinline__ void tdm_load_2d(const __bf16* gptr,
                                                   unsigned lds_byte_off,
                                                   int rows, int cols,
                                                   int row_stride_elems) {
  unsigned long long ga = (unsigned long long)(uintptr_t)gptr;
  u32x4 g0;
  g0[0] = 1u;                                              // count=1
  g0[1] = lds_byte_off;                                    // lds_addr
  g0[2] = (unsigned)(ga & 0xFFFFFFFFu);                    // global_addr[31:0]
  g0[3] = (unsigned)((ga >> 32) & 0x01FFFFFFu) | (2u << 30); // addr[56:32] | type=2
  const unsigned td0 = 0x40000000u;                        // huge tensor_dim0
  const unsigned td1 = 0x40000000u;                        // huge tensor_dim1
  i32x8 g1;
  g1[0] = (int)(1u << 16);                                 // data_size=1 (2 bytes)
  g1[1] = (int)((td0 & 0xFFFFu) << 16);                    // tensor_dim0 lo
  g1[2] = (int)((td0 >> 16) | ((td1 & 0xFFFFu) << 16));    // dim0 hi | dim1 lo
  g1[3] = (int)((td1 >> 16) | ((unsigned)cols << 16));     // dim1 hi | tile_dim0
  g1[4] = rows;                                            // tile_dim1 (tile_dim2=0)
  g1[5] = row_stride_elems;                                // tensor_dim0_stride lo
  g1[6] = 0;                                               // stride hi | dim1_stride lo
  g1[7] = 0;
  i32x4 z4 = {0, 0, 0, 0};
  i32x8 z8 = {0, 0, 0, 0, 0, 0, 0, 0};
  __builtin_amdgcn_tensor_load_to_lds(g0, g1, z4, z4, z8, 0);
}
#endif

// ---------------- elementwise f32 -> bf16 ----------------
__global__ __launch_bounds__(256) void cvt_bf16_kernel(const float* __restrict__ in,
                                                       __bf16* __restrict__ out, int n) {
  int i = blockIdx.x * 256 + threadIdx.x;
  if (i < n) out[i] = (__bf16)in[i];
}

// w2 [NC,HID] -> bf16 padded to [NPAD,HID] with zero rows
__global__ __launch_bounds__(256) void cvt_w2_kernel(const float* __restrict__ in,
                                                     __bf16* __restrict__ out) {
  int i = blockIdx.x * 256 + threadIdx.x;
  if (i >= NPAD * HID) return;
  int r = i >> 11;                       // / HID
  int c = i & (HID - 1);
  out[i] = (r < NC) ? (__bf16)in[(size_t)r * HID + c] : (__bf16)0.0f;
}

// ---------------- row LayerNorm -> bf16 (one wave per row of 512) -------------
__global__ __launch_bounds__(256) void ln_rows_kernel(const float* __restrict__ x,
                                                      const float* __restrict__ g,
                                                      const float* __restrict__ b,
                                                      __bf16* __restrict__ out, int R) {
  int row  = blockIdx.x * 8 + (threadIdx.x >> 5);
  if (row >= R) return;                               // wave-uniform
  int lane = threadIdx.x & 31;
  const float* xr = x + (size_t)row * DIM;
  float v[16], s = 0.f, s2 = 0.f;
#pragma unroll
  for (int j = 0; j < 16; j++) { float t = xr[lane + 32 * j]; v[j] = t; s += t; s2 += t * t; }
#pragma unroll
  for (int o = 16; o >= 1; o >>= 1) { s += __shfl_xor(s, o, 32); s2 += __shfl_xor(s2, o, 32); }
  float m   = s * (1.0f / DIM);
  float var = s2 * (1.0f / DIM) - m * m;
  float inv = rsqrtf(var + 1e-5f);
  __bf16* orow = out + (size_t)row * DIM;
#pragma unroll
  for (int j = 0; j < 16; j++) {
    int c = lane + 32 * j;
    orow[c] = (__bf16)((v[j] - m) * inv * g[c] + b[c]);
  }
}

// ---------------- row l2-normalize -> bf16 ----------------
__global__ __launch_bounds__(256) void l2n_rows_kernel(const float* __restrict__ x,
                                                       __bf16* __restrict__ out, int R) {
  int row  = blockIdx.x * 8 + (threadIdx.x >> 5);
  if (row >= R) return;
  int lane = threadIdx.x & 31;
  const float* xr = x + (size_t)row * DIM;
  float v[16], s2 = 0.f;
#pragma unroll
  for (int j = 0; j < 16; j++) { float t = xr[lane + 32 * j]; v[j] = t; s2 += t * t; }
#pragma unroll
  for (int o = 16; o >= 1; o >>= 1) s2 += __shfl_xor(s2, o, 32);
  float inv = 1.0f / fmaxf(sqrtf(s2), 1e-12f);
  __bf16* orow = out + (size_t)row * DIM;
#pragma unroll
  for (int j = 0; j < 16; j++) orow[lane + 32 * j] = (__bf16)(v[j] * inv);
}

// ---------------- generic bf16 WMMA GEMM:  C = act(alpha*A@Bt^T + bias) -------
// A:[M,K] bf16, Bt:[N,K] bf16. Block = 8 waves; wave computes 16 rows x 64 cols.
// REQUIRES: M % 128 == 0, N % 64 == 0, K % KC == 0 (all call sites satisfy this).
// B tile (64 x KC) is staged into LDS by the Tensor Data Mover, double-buffered:
// each wave DMAs its own 8-row slice, waits TENSORcnt, block barrier, computes
// chunk c from LDS while chunk c+1 streams in.  All fragments of one K-step are
// loaded before the 4 back-to-back WMMAs so DS loads stay in flight.
__global__ __launch_bounds__(256) void gemm_bf16_kernel(
    const __bf16* __restrict__ A, const __bf16* __restrict__ Bt,
    __bf16* Cb, float* Cf, const float* __restrict__ bias,
    int M, int N, int Nvalid, int K, float alpha, int relu, int ldc,
    long long strideA, long long strideB, long long strideC) {
  int batch = blockIdx.z;
  A  += (size_t)batch * strideA;
  Bt += (size_t)batch * strideB;
  if (Cb) Cb += (size_t)batch * strideC;
  if (Cf) Cf += (size_t)batch * strideC;

  int wave = threadIdx.x >> 5;
  int lane = threadIdx.x & 31;
  int m0 = blockIdx.y * 128 + wave * 16;
  int n0 = blockIdx.x * 64;

  v8f acc[4] = {v8f{}, v8f{}, v8f{}, v8f{}};

#if USE_TDM
  __shared__ __align__(16) __bf16 Bs[2][64 * KC];      // 32 KB double buffer
  const __bf16* Bblk = Bt + (size_t)n0 * K;            // 64 rows x K
  int nchunks = K / KC;
  // issue chunk 0: wave w DMAs rows [w*8, w*8+8) of the 64-row tile
  tdm_load_2d(Bblk + (size_t)(wave * 8) * K,
              (unsigned)(uintptr_t)&Bs[0][(wave * 8) * KC], 8, KC, K);
  for (int c = 0; c < nchunks; ++c) {
    int buf = c & 1;
    __builtin_amdgcn_s_wait_tensorcnt(0);              // our slice of chunk c done
    __syncthreads();                                   // everyone's slices done
    if (c + 1 < nchunks)                               // overlap DMA of c+1 with compute of c
      tdm_load_2d(Bblk + (size_t)(wave * 8) * K + (size_t)(c + 1) * KC,
                  (unsigned)(uintptr_t)&Bs[buf ^ 1][(wave * 8) * KC], 8, KC, K);
    const __bf16* Ab = A + (size_t)m0 * K + (size_t)c * KC;
    for (int kk = 0; kk < KC; kk += 32) {
      v16bf a = load_a_frag(Ab + kk, K);
      v16bf bf[4];
#pragma unroll
      for (int t = 0; t < 4; t++)
        bf[t] = load_b_frag(&Bs[buf][(t * 16) * KC + kk], KC);
#pragma unroll
      for (int t = 0; t < 4; t++)
        acc[t] = __builtin_amdgcn_wmma_f32_16x16x32_bf16(false, a, false, bf[t],
                                                         (short)0, acc[t], false, false);
    }
    __syncthreads();                                   // done reading buf before reuse
  }
#else
  for (int k0 = 0; k0 < K; k0 += 32) {
    v16bf a = load_a_frag(A + (size_t)m0 * K + k0, K);
    v16bf bf[4];
#pragma unroll
    for (int t = 0; t < 4; t++)
      bf[t] = load_b_frag(Bt + (size_t)(n0 + t * 16) * K + k0, K);
#pragma unroll
    for (int t = 0; t < 4; t++)
      acc[t] = __builtin_amdgcn_wmma_f32_16x16x32_bf16(false, a, false, bf[t],
                                                       (short)0, acc[t], false, false);
  }
#endif

  int colL = lane & 15;
  int rb   = (lane >> 4) << 3;                        // 0 or 8
#pragma unroll
  for (int t = 0; t < 4; t++) {
    int col = n0 + t * 16 + colL;
    if (col >= Nvalid) continue;
    float bv = bias ? bias[col] : 0.0f;
#pragma unroll
    for (int j = 0; j < 8; j++) {
      int row = m0 + rb + j;
      float val = acc[t][j] * alpha + bv;
      if (relu) val = fmaxf(val, 0.0f);
      if (Cb) Cb[(size_t)row * ldc + col] = (__bf16)val;
      else    Cf[(size_t)row * ldc + col] = val;
    }
  }
}

// ---------------- fused attention + x2 kernel --------------------------------
// grid = (BQ/16, BKB), block = 128 (4 waves). Per block: 16 queries x 1 key-batch.
// S = scale * qb @ kb^T      [16,128]   (WMMA)
// P = qxn @ kx^T             [16,128]   (WMMA)
// a = softmax(mask(S))                  (VALU)
// num = sum a*P ; den^2 = a^T G_k a     (WMMA for a@G_k)
// x2 = exp(ls) * num / max(den,1e-12)
__global__ __launch_bounds__(128) void attn_kernel(
    const __bf16* __restrict__ qb, const __bf16* __restrict__ qnb,
    const __bf16* __restrict__ kb, const __bf16* __restrict__ kxb,
    const __bf16* __restrict__ Gb, const unsigned char* __restrict__ mask,
    const float* __restrict__ logit_scale, float* __restrict__ x2) {
  int qt = blockIdx.x;                 // query tile (16 rows)
  int kB = blockIdx.y;                 // key-batch
  __shared__ __align__(16) float  S[16][NKEY];
  __shared__ __align__(16) float  P[16][NKEY];
  __shared__ __align__(16) __bf16 Abf[16][NKEY];
  __shared__ float num[16];

  int tid = threadIdx.x, wave = tid >> 5, lane = tid & 31;
  int n0  = wave * 32;

  const __bf16* Aq = qb  + (size_t)(qt * 16) * DIM;
  const __bf16* An = qnb + (size_t)(qt * 16) * DIM;
  const __bf16* Bk = kb  + (size_t)(kB * NKEY) * DIM;
  const __bf16* Bx = kxb + (size_t)(kB * NKEY) * DIM;

  v8f s0{}, s1{}, p0{}, p1{};
  for (int k0 = 0; k0 < DIM; k0 += 32) {
    v16bf aq = load_a_frag(Aq + k0, DIM);
    v16bf an = load_a_frag(An + k0, DIM);
    v16bf b0 = load_b_frag(Bk + (size_t)(n0)      * DIM + k0, DIM);
    v16bf b1 = load_b_frag(Bk + (size_t)(n0 + 16) * DIM + k0, DIM);
    v16bf c0 = load_b_frag(Bx + (size_t)(n0)      * DIM + k0, DIM);
    v16bf c1 = load_b_frag(Bx + (size_t)(n0 + 16) * DIM + k0, DIM);
    s0 = __builtin_amdgcn_wmma_f32_16x16x32_bf16(false, aq, false, b0, (short)0, s0, false, false);
    s1 = __builtin_amdgcn_wmma_f32_16x16x32_bf16(false, aq, false, b1, (short)0, s1, false, false);
    p0 = __builtin_amdgcn_wmma_f32_16x16x32_bf16(false, an, false, c0, (short)0, p0, false, false);
    p1 = __builtin_amdgcn_wmma_f32_16x16x32_bf16(false, an, false, c1, (short)0, p1, false, false);
  }
  const float scale = 0.044194173824159216f;          // 512^-0.5
  int colL = lane & 15, rb = (lane >> 4) << 3;
#pragma unroll
  for (int j = 0; j < 8; j++) {
    S[rb + j][n0 + colL]      = s0[j] * scale;
    S[rb + j][n0 + 16 + colL] = s1[j] * scale;
    P[rb + j][n0 + colL]      = p0[j];
    P[rb + j][n0 + 16 + colL] = p1[j];
  }
  __syncthreads();

  // masked softmax + numerator (16 rows handled by threads 0..15)
  if (tid < 16) {
    int r = tid;
    const unsigned char* m = mask + (size_t)kB * NKEY;
    float mx = -INFINITY;
    for (int n = 0; n < NKEY; n++) if (!m[n]) mx = fmaxf(mx, S[r][n]);
    float sum = 0.f;
    for (int n = 0; n < NKEY; n++) {
      float e = m[n] ? 0.0f : __expf(S[r][n] - mx);
      S[r][n] = e; sum += e;
    }
    float inv = 1.0f / sum;
    float acc = 0.f;
    for (int n = 0; n < NKEY; n++) {
      float a = S[r][n] * inv;
      S[r][n]   = a;
      Abf[r][n] = (__bf16)a;
      acc += a * P[r][n];
    }
    num[r] = acc;
  }
  __syncthreads();

  // T = a @ G_k  (G symmetric, bf16 [128,128]); write into P (safe: phase2 done)
  const __bf16* G = Gb + (size_t)kB * NKEY * NKEY;
  v8f t0{}, t1{};
  for (int k0 = 0; k0 < NKEY; k0 += 32) {
    v16bf a  = load_a_frag(&Abf[0][0] + k0, NKEY);
    v16bf g0 = load_b_frag(G + (size_t)(n0)      * NKEY + k0, NKEY);
    v16bf g1 = load_b_frag(G + (size_t)(n0 + 16) * NKEY + k0, NKEY);
    t0 = __builtin_amdgcn_wmma_f32_16x16x32_bf16(false, a, false, g0, (short)0, t0, false, false);
    t1 = __builtin_amdgcn_wmma_f32_16x16x32_bf16(false, a, false, g1, (short)0, t1, false, false);
  }
#pragma unroll
  for (int j = 0; j < 8; j++) {
    P[rb + j][n0 + colL]      = t0[j];
    P[rb + j][n0 + 16 + colL] = t1[j];
  }
  __syncthreads();

  if (tid < 16) {
    int r = tid;
    float d2 = 0.f;
    for (int n = 0; n < NKEY; n++) d2 += S[r][n] * P[r][n];
    float den = sqrtf(fmaxf(d2, 0.0f));
    float ls  = expf(logit_scale[0]);
    x2[(size_t)(qt * 16 + r) * BKB + kB] = ls * num[r] / fmaxf(den, 1e-12f);
  }
}

// =============================== launcher ====================================
extern "C" void kernel_launch(void* const* d_in, const int* in_sizes, int n_in,
                              void* d_out, int out_size, void* d_ws, size_t ws_size,
                              hipStream_t stream) {
  const float* qx = (const float*)d_in[0];
  const float* kx = (const float*)d_in[1];
  const unsigned char* mask = (const unsigned char*)d_in[2];   // numpy bool8
  const float* logit_scale = (const float*)d_in[3];
  const float* g_q = (const float*)d_in[4];
  const float* b_q = (const float*)d_in[5];
  const float* g_k = (const float*)d_in[6];
  const float* b_k = (const float*)d_in[7];
  const float* wq = (const float*)d_in[8];
  const float* wk = (const float*)d_in[9];
  const float* w1 = (const float*)d_in[10];
  const float* b1 = (const float*)d_in[11];
  const float* w2 = (const float*)d_in[12];
  const float* b2 = (const float*)d_in[13];

  float* x1 = (float*)d_out;                 // [BQ, NC]
  float* x2 = x1 + (size_t)BQ * NC;          // [BQ, BKB]

  // workspace layout (bf16 staging, ~115 MB total)
  char* ws = (char*)d_ws;
  size_t off = 0;
  auto alloc = [&](size_t bytes) { char* p = ws + off; off += (bytes + 255) & ~(size_t)255; return p; };
  const int RK = BKB * NKEY;                               // 32768 key rows
  __bf16* qxb  = (__bf16*)alloc((size_t)BQ * DIM * 2);     // bf16(qx)
  __bf16* qlnb = (__bf16*)alloc((size_t)BQ * DIM * 2);     // bf16(LN(qx))
  __bf16* qnb  = (__bf16*)alloc((size_t)BQ * DIM * 2);     // bf16(l2norm(qx))
  __bf16* qb   = (__bf16*)alloc((size_t)BQ * DIM * 2);     // LN(qx)@wq^T
  __bf16* wqb  = (__bf16*)alloc((size_t)DIM * DIM * 2);
  __bf16* wkb  = (__bf16*)alloc((size_t)DIM * DIM * 2);
  __bf16* w1b  = (__bf16*)alloc((size_t)HID * DIM * 2);
  __bf16* w2b  = (__bf16*)alloc((size_t)NPAD * HID * 2);   // zero-padded rows
  __bf16* hb   = (__bf16*)alloc((size_t)BQ * HID * 2);     // relu(qx@w1^T+b1)
  __bf16* klnb = (__bf16*)alloc((size_t)RK * DIM * 2);     // bf16(LN(kx))
  __bf16* kb   = (__bf16*)alloc((size_t)RK * DIM * 2);     // LN(kx)@wk^T
  __bf16* kxb  = (__bf16*)alloc((size_t)RK * DIM * 2);     // bf16(kx)
  __bf16* Gb   = (__bf16*)alloc((size_t)BKB * NKEY * NKEY * 2); // grams
  (void)ws_size; (void)in_sizes; (void)n_in; (void)out_size;

  // --- stage 0: conversions ---
  cvt_bf16_kernel<<<(BQ * DIM + 255) / 256, 256, 0, stream>>>(qx, qxb, BQ * DIM);
  cvt_bf16_kernel<<<(DIM * DIM + 255) / 256, 256, 0, stream>>>(wq, wqb, DIM * DIM);
  cvt_bf16_kernel<<<(DIM * DIM + 255) / 256, 256, 0, stream>>>(wk, wkb, DIM * DIM);
  cvt_bf16_kernel<<<(HID * DIM + 255) / 256, 256, 0, stream>>>(w1, w1b, HID * DIM);
  cvt_bf16_kernel<<<(RK * DIM + 255) / 256, 256, 0, stream>>>(kx, kxb, RK * DIM);
  cvt_w2_kernel<<<(NPAD * HID + 255) / 256, 256, 0, stream>>>(w2, w2b);

  // --- stage 1: row normalizations ---
  ln_rows_kernel<<<BQ / 8, 256, 0, stream>>>(qx, g_q, b_q, qlnb, BQ);
  ln_rows_kernel<<<RK / 8, 256, 0, stream>>>(kx, g_k, b_k, klnb, RK);
  l2n_rows_kernel<<<BQ / 8, 256, 0, stream>>>(qx, qnb, BQ);

  // --- stage 2: projections (WMMA GEMMs, TDM-staged B tiles) ---
  gemm_bf16_kernel<<<dim3(DIM / 64, BQ / 128, 1), 256, 0, stream>>>(
      qlnb, wqb, qb, nullptr, nullptr, BQ, DIM, DIM, DIM, 1.0f, 0, DIM, 0, 0, 0);
  gemm_bf16_kernel<<<dim3(DIM / 64, RK / 128, 1), 256, 0, stream>>>(
      klnb, wkb, kb, nullptr, nullptr, RK, DIM, DIM, DIM, 1.0f, 0, DIM, 0, 0, 0);

  // --- stage 3: per-key-batch Gram matrices G_k = kx_k @ kx_k^T ---
  gemm_bf16_kernel<<<dim3(NKEY / 64, 1, BKB), 256, 0, stream>>>(
      kxb, kxb, Gb, nullptr, nullptr, NKEY, NKEY, NKEY, DIM, 1.0f, 0, NKEY,
      (long long)NKEY * DIM, (long long)NKEY * DIM, (long long)NKEY * NKEY);

  // --- stage 4: MLP head (x1) ---
  gemm_bf16_kernel<<<dim3(HID / 64, BQ / 128, 1), 256, 0, stream>>>(
      qxb, w1b, hb, nullptr, b1, BQ, HID, HID, DIM, 1.0f, 1, HID, 0, 0, 0);
  gemm_bf16_kernel<<<dim3(NPAD / 64, BQ / 128, 1), 256, 0, stream>>>(
      hb, w2b, nullptr, x1, b2, BQ, NPAD, NC, HID, 1.0f, 0, NC, 0, 0, 0);

  // --- stage 5: fused attention + similarity head (x2) ---
  attn_kernel<<<dim3(BQ / 16, BKB), 128, 0, stream>>>(
      qb, qnb, kb, kxb, Gb, mask, logit_scale, x2);
}
